// TextGenerationModel_73237782331782
// MI455X (gfx1250) — matrix operand order repaired and checked by reading, not compiled, not run on hardware
//
#include <hip/hip_runtime.h>
#include <hip/hip_bf16.h>

// ---------- types / helpers ----------
typedef _Float16 half8  __attribute__((ext_vector_type(8)));
typedef _Float16 half16 __attribute__((ext_vector_type(16)));
typedef float    float8 __attribute__((ext_vector_type(8)));

__device__ __forceinline__ float8 wmma_f16(half16 a, half16 b, float8 c) {
  // D = A(16x32 f16) * B(32x16 f16) + C(16x16 f32)
  return __builtin_amdgcn_wmma_f32_16x16x32_f16(false, a, false, b, (short)0, c,
                                                false, false);
}
__device__ __forceinline__ half16 cat16(half8 lo, half8 hi) {
  return __builtin_shufflevector(lo, hi, 0,1,2,3,4,5,6,7,8,9,10,11,12,13,14,15);
}
__device__ __forceinline__ float sigmf(float x) { return 1.0f / (1.0f + __expf(-x)); }

#define BSZ   512
#define TLEN  128
#define EMBD  128
#define NVOC  50000

// ---------- weight transpose + fp16 convert: W[K,N] f32 -> WT[N,K] f16 ----------
__global__ void cvt_transpose(const float* __restrict__ W, _Float16* __restrict__ WT,
                              int K, int N) {
  int i = blockIdx.x * blockDim.x + threadIdx.x;       // i indexes WT: i = n*K + k
  if (i >= K * N) return;
  int n = i / K, k = i - n * K;
  WT[i] = (_Float16)W[(size_t)k * N + n];
}

// ---------- embedding gather -> fp16 activations [B*T, 128] ----------
__global__ void gather_emb(const int* __restrict__ x, const float* __restrict__ emb,
                           _Float16* __restrict__ out) {
  size_t i = (size_t)blockIdx.x * blockDim.x + threadIdx.x;  // over B*T*128
  if (i >= (size_t)BSZ * TLEN * EMBD) return;
  int tok = x[i >> 7];
  out[i] = (_Float16)emb[(size_t)tok * EMBD + (i & 127)];
}

// ---------- generic WMMA GEMM: C[M,N] = A[M,K]f16 * BT[N,K]f16^T + bias[N] ----------
// One wave computes NTPW consecutive 16x16 tiles (A fragment reused across them).
template <int OUT_F16, int KDIM, int NTPW>
__global__ void gemm_wmma(const _Float16* __restrict__ A, const _Float16* __restrict__ BT,
                          const float* __restrict__ bias, void* __restrict__ Cv,
                          int M, int N) {
  const int lane = threadIdx.x & 31;
  const int wave = threadIdx.x >> 5;
  const int gw   = blockIdx.x * (blockDim.x >> 5) + wave;
  const int nTiles = N >> 4, mTiles = M >> 4;
  const int nGroups = nTiles / NTPW;
  if (gw >= mTiles * nGroups) return;                  // wave-uniform
  const int mTile  = gw / nGroups;
  const int nTile0 = (gw - mTile * nGroups) * NTPW;
  const int l15 = lane & 15, hi = lane >> 4;

  // A fragment: lane row = mTile*16 + l15 ; 16 contiguous K at k0+hi*16
  const _Float16* arow = A + (size_t)(mTile * 16 + l15) * KDIM + hi * 16;
  // B fragments (BT is [N,K]): lane col = tile*16 + l15 ; 16 contiguous K at k0+hi*16
  const _Float16* brow[NTPW];
#pragma unroll
  for (int j = 0; j < NTPW; ++j)
    brow[j] = BT + (size_t)((nTile0 + j) * 16 + l15) * KDIM + hi * 16;

  float8 acc[NTPW];
#pragma unroll
  for (int j = 0; j < NTPW; ++j) acc[j] = (float8){};

#pragma unroll
  for (int k0 = 0; k0 < KDIM; k0 += 32) {
    half16 af = cat16(*(const half8*)(arow + k0), *(const half8*)(arow + k0 + 8));
#pragma unroll
    for (int j = 0; j < NTPW; ++j) {
      half16 bf = *(const half16*)(brow[j] + k0);
      acc[j] = wmma_f16(af, bf, acc[j]);
    }
  }
  const int m0 = mTile * 16 + hi * 8;
#pragma unroll
  for (int j = 0; j < NTPW; ++j) {
    const int n  = (nTile0 + j) * 16 + l15;
    const float bv = bias[n];
    if (OUT_F16) {
      _Float16* C = (_Float16*)Cv;
#pragma unroll
      for (int i = 0; i < 8; ++i) C[(size_t)(m0 + i) * N + n] = (_Float16)(acc[j][i] + bv);
    } else {
      float* C = (float*)Cv;
#pragma unroll
      for (int i = 0; i < 8; ++i) C[(size_t)(m0 + i) * N + n] = acc[j][i] + bv;
    }
  }
}

// ---------- GRU recurrence (one direction, one layer), U compile-time ----------
// Workgroup owns 16 batch rows for all T steps. waves = 3U/16, blockDim = 6U.
// XZ = x@W + bi, fp16 [B*T, 3U]. UT = U^T fp16 [3U, U]. rec = h@U + br via WMMA.
template <int U>
__global__ void gru_rec(const _Float16* __restrict__ XZ, const _Float16* __restrict__ UT,
                        const float* __restrict__ br, _Float16* __restrict__ Hout,
                        int ostride, int ocol, int reverse, int finalOnly) {
  constexpr int N3   = 3 * U;
  constexpr int HSTR = 72;                  // padded LDS row (fits U=64, 16B-aligned)
  const int b0   = blockIdx.x * 16;
  const int lane = threadIdx.x & 31;
  const int wave = threadIdx.x >> 5;        // n-tile index, 0..3U/16-1
  const int l15  = lane & 15, hi = lane >> 4;

  __shared__ _Float16 hbuf[2][16][HSTR];    // double-buffered state
  __shared__ float    recs[16][N3 + 8];     // rec = h@U + br

  // Resident B fragments of U^T for this wave's 16 output columns.
  const _Float16* bp = UT + (size_t)(wave * 16 + l15) * U + hi * 16;
  half16 bf0 = *(const half16*)(bp);
  half16 bf1 = {};
  if constexpr (U > 32) bf1 = *(const half16*)(bp + 32);

  for (int e = threadIdx.x; e < 2 * 16 * HSTR; e += 6 * U)
    ((_Float16*)hbuf)[e] = (_Float16)0.0f;
  __syncthreads();

  int cur = 0;
  for (int t = 0; t < TLEN; ++t) {
    const int tt = reverse ? (TLEN - 1 - t) : t;

    // rec tile via WMMA from LDS state
    float8 acc = {};
    const _Float16* hp = &hbuf[cur][l15][hi * 16];
    half16 af = cat16(*(const half8*)(hp), *(const half8*)(hp + 8));
    acc = wmma_f16(af, bf0, acc);
    if constexpr (U > 32) {
      af = cat16(*(const half8*)(hp + 32), *(const half8*)(hp + 40));
      acc = wmma_f16(af, bf1, acc);
    }
    const float bv = br[wave * 16 + l15];
#pragma unroll
    for (int i = 0; i < 8; ++i) recs[i + hi * 8][wave * 16 + l15] = acc[i] + bv;
    __syncthreads();

    // gates: z,r in cols [0,U),[U,2U); candidate uses cols [2U,3U)
    for (int e = threadIdx.x; e < 16 * U; e += 6 * U) {
      const int row = e / U, j = e & (U - 1);          // shifts/masks: U is 2^k
      const size_t xb = ((size_t)(b0 + row) * TLEN + tt) * (size_t)N3;
      const float z  = sigmf((float)XZ[xb + j]     + recs[row][j]);
      const float rg = sigmf((float)XZ[xb + U + j] + recs[row][U + j]);
      float hh = (float)XZ[xb + 2 * U + j] + rg * recs[row][2 * U + j];
      hh = hh > 0.0f ? hh : 0.0f;                      // relu activation
      const float hold = (float)hbuf[cur][row][j];
      const float hn = z * hold + (1.0f - z) * hh;
      hbuf[cur ^ 1][row][j] = (_Float16)hn;
      if (!finalOnly)
        Hout[((size_t)(b0 + row) * TLEN + tt) * ostride + ocol + j] = (_Float16)hn;
    }
    __syncthreads();
    cur ^= 1;
  }
  if (finalOnly) {
    for (int e = threadIdx.x; e < 16 * U; e += 6 * U) {
      const int row = e / U, j = e & (U - 1);
      Hout[(size_t)(b0 + row) * ostride + ocol + j] = hbuf[cur][row][j];
    }
  }
}

// ---------- row softmax over d_out [512, 50000] in place ----------
__global__ void softmax_rows(float* __restrict__ out, int N) {
  __shared__ float red[256];
  float* rowp = out + (size_t)blockIdx.x * N;
  float m = -1e30f;
  for (int i = threadIdx.x; i < N; i += blockDim.x) {
    __builtin_prefetch(rowp + i + 4096, 0, 1);         // global_prefetch_b8
    m = fmaxf(m, rowp[i]);
  }
  red[threadIdx.x] = m; __syncthreads();
  for (int s = 128; s > 0; s >>= 1) {
    if (threadIdx.x < s) red[threadIdx.x] = fmaxf(red[threadIdx.x], red[threadIdx.x + s]);
    __syncthreads();
  }
  m = red[0]; __syncthreads();
  float sum = 0.0f;
  for (int i = threadIdx.x; i < N; i += blockDim.x) {
    __builtin_prefetch(rowp + i + 4096, 0, 1);
    sum += __expf(rowp[i] - m);
  }
  red[threadIdx.x] = sum; __syncthreads();
  for (int s = 128; s > 0; s >>= 1) {
    if (threadIdx.x < s) red[threadIdx.x] += red[threadIdx.x + s];
    __syncthreads();
  }
  const float inv = 1.0f / red[0];
  for (int i = threadIdx.x; i < N; i += blockDim.x) rowp[i] = __expf(rowp[i] - m) * inv;
}

// ---------- host orchestration ----------
extern "C" void kernel_launch(void* const* d_in, const int* in_sizes, int n_in,
                              void* d_out, int out_size, void* d_ws, size_t ws_size,
                              hipStream_t stream) {
  (void)in_sizes; (void)n_in; (void)out_size; (void)ws_size;
  const int*   x    = (const int*)  d_in[0];
  const float* emb  = (const float*)d_in[1];
  const float* W1f = (const float*)d_in[2],  *U1f = (const float*)d_in[3];
  const float* bi1f= (const float*)d_in[4],  *br1f= (const float*)d_in[5];
  const float* W1b = (const float*)d_in[6],  *U1b = (const float*)d_in[7];
  const float* bi1b= (const float*)d_in[8],  *br1b= (const float*)d_in[9];
  const float* W2f = (const float*)d_in[10], *U2f = (const float*)d_in[11];
  const float* bi2f= (const float*)d_in[12], *br2f= (const float*)d_in[13];
  const float* W2b = (const float*)d_in[14], *U2b = (const float*)d_in[15];
  const float* bi2b= (const float*)d_in[16], *br2b= (const float*)d_in[17];
  const float* W3f = (const float*)d_in[18], *U3f = (const float*)d_in[19];
  const float* bi3f= (const float*)d_in[20], *br3f= (const float*)d_in[21];
  const float* W3b = (const float*)d_in[22], *U3b = (const float*)d_in[23];
  const float* bi3b= (const float*)d_in[24], *br3b= (const float*)d_in[25];
  const float* Wd  = (const float*)d_in[26], *bd  = (const float*)d_in[27];

  char* ws = (char*)d_ws;
  size_t off = 0;
  auto take = [&](size_t bytes) { size_t r = off; off = (off + bytes + 255) & ~(size_t)255; return r; };

  // converted weights (fp16, transposed to [N,K])
  const size_t oW1fT = take(192*128*2), oU1fT = take(192*64*2);
  const size_t oW1bT = take(192*128*2), oU1bT = take(192*64*2);
  const size_t oW2fT = take(96*128*2),  oU2fT = take(96*32*2);
  const size_t oW2bT = take(96*128*2),  oU2bT = take(96*32*2);
  const size_t oW3fT = take(96*64*2),   oU3fT = take(96*32*2);
  const size_t oW3bT = take(96*64*2),   oU3bT = take(96*32*2);
  const size_t oWdT  = take((size_t)NVOC*64*2);
  // activations (region-reused)
  const size_t BT = (size_t)BSZ * TLEN;
  const size_t oA0 = take(BT * 128 * 2);       // emb fp16; region reused for H2/Hfin
  const size_t oH1 = take(BT * 128 * 2);       // layer1 concat output
  const size_t oXZ = take(BT * 192 * 2);       // shared xz buffer (max 3*64 cols)
  const size_t oH2   = oA0;                    // [BT,64] fp16, reuses dead A0 region
  const size_t oHfin = oA0 + BT * 64 * 2;      // [512,64] fp16

  auto cvt = [&](const float* src, size_t dst, int K, int N) {
    int total = K * N;
    cvt_transpose<<<(total + 255) / 256, 256, 0, stream>>>(src, (_Float16*)(ws + dst), K, N);
  };
  cvt(W1f, oW1fT, 128, 192); cvt(U1f, oU1fT, 64, 192);
  cvt(W1b, oW1bT, 128, 192); cvt(U1b, oU1bT, 64, 192);
  cvt(W2f, oW2fT, 128, 96);  cvt(U2f, oU2fT, 32, 96);
  cvt(W2b, oW2bT, 128, 96);  cvt(U2b, oU2bT, 32, 96);
  cvt(W3f, oW3fT, 64, 96);   cvt(U3f, oU3fT, 32, 96);
  cvt(W3b, oW3bT, 64, 96);   cvt(U3b, oU3bT, 32, 96);
  cvt(Wd,  oWdT,  64, NVOC);

  gather_emb<<<(int)((BT * 128 + 255) / 256), 256, 0, stream>>>(x, emb, (_Float16*)(ws + oA0));

  // projection GEMMs: fp16 out, 2 N-tiles per wave, 8 waves per block
  auto gemmK128 = [&](size_t aOff, size_t btOff, const float* bias, size_t cOff, int M, int N) {
    int blocks = (M / 16) * ((N / 16) / 2) / 8;
    gemm_wmma<1, 128, 2><<<blocks, 256, 0, stream>>>((const _Float16*)(ws + aOff),
                                                     (const _Float16*)(ws + btOff), bias,
                                                     (void*)(ws + cOff), M, N);
  };
  auto gemmK64 = [&](size_t aOff, size_t btOff, const float* bias, size_t cOff, int M, int N) {
    int blocks = (M / 16) * ((N / 16) / 2) / 8;
    gemm_wmma<1, 64, 2><<<blocks, 256, 0, stream>>>((const _Float16*)(ws + aOff),
                                                    (const _Float16*)(ws + btOff), bias,
                                                    (void*)(ws + cOff), M, N);
  };

  // layer 1 (U=64): input A0 [BT,128]
  gemmK128(oA0, oW1fT, bi1f, oXZ, (int)BT, 192);
  gru_rec<64><<<BSZ / 16, 384, 0, stream>>>((const _Float16*)(ws + oXZ),
      (const _Float16*)(ws + oU1fT), br1f, (_Float16*)(ws + oH1), 128, 0, 0, 0);
  gemmK128(oA0, oW1bT, bi1b, oXZ, (int)BT, 192);
  gru_rec<64><<<BSZ / 16, 384, 0, stream>>>((const _Float16*)(ws + oXZ),
      (const _Float16*)(ws + oU1bT), br1b, (_Float16*)(ws + oH1), 128, 64, 1, 0);
  // layer 2 (U=32): input H1 [BT,128]
  gemmK128(oH1, oW2fT, bi2f, oXZ, (int)BT, 96);
  gru_rec<32><<<BSZ / 16, 192, 0, stream>>>((const _Float16*)(ws + oXZ),
      (const _Float16*)(ws + oU2fT), br2f, (_Float16*)(ws + oH2), 64, 0, 0, 0);
  gemmK128(oH1, oW2bT, bi2b, oXZ, (int)BT, 96);
  gru_rec<32><<<BSZ / 16, 192, 0, stream>>>((const _Float16*)(ws + oXZ),
      (const _Float16*)(ws + oU2bT), br2b, (_Float16*)(ws + oH2), 64, 32, 1, 0);
  // layer 3 (U=32): input H2 [BT,64]; only final states kept
  gemmK64(oH2, oW3fT, bi3f, oXZ, (int)BT, 96);
  gru_rec<32><<<BSZ / 16, 192, 0, stream>>>((const _Float16*)(ws + oXZ),
      (const _Float16*)(ws + oU3fT), br3f, (_Float16*)(ws + oHfin), 64, 0, 0, 1);
  gemmK64(oH2, oW3bT, bi3b, oXZ, (int)BT, 96);
  gru_rec<32><<<BSZ / 16, 192, 0, stream>>>((const _Float16*)(ws + oXZ),
      (const _Float16*)(ws + oU3bT), br3b, (_Float16*)(ws + oHfin), 64, 32, 1, 1);

  // dense head: [512,64] x [64,50000] + bd -> f32 logits in d_out
  {
    // N tiles = 3125 = 5^5 ; 5 tiles per wave -> 625 groups ; 32*625/8 = 2500 blocks
    int blocks = (BSZ / 16) * ((NVOC / 16) / 5) / 8;
    gemm_wmma<0, 64, 5><<<blocks, 256, 0, stream>>>((const _Float16*)(ws + oHfin),
                                                    (const _Float16*)(ws + oWdT), bd,
                                                    d_out, BSZ, NVOC);
  }
  softmax_rows<<<BSZ, 256, 0, stream>>>((float*)d_out, NVOC);
}
